// EGNNEncoder_79714593014003
// MI455X (gfx1250) — matrix operand order repaired and checked by reading, not compiled, not run on hardware
//
#include <hip/hip_runtime.h>
#include <hip/hip_bf16.h>

#define HD 128
#define NB 16000
#define NA 80000
#define NE 320000
#define NG 128
#define NL 3

typedef __attribute__((ext_vector_type(16))) __bf16 v16bf;
typedef __attribute__((ext_vector_type(8)))  float  v8f;

struct U8x32 { unsigned int u[8]; };

__device__ __forceinline__ unsigned short f2bf(float f) {
    unsigned int u = __builtin_bit_cast(unsigned int, f);
    u += 0x7fffu + ((u >> 16) & 1u);            // round-to-nearest-even
    return (unsigned short)(u >> 16);
}
__device__ __forceinline__ float bf2f(unsigned short h) {
    unsigned int u = ((unsigned int)h) << 16;
    return __builtin_bit_cast(float, u);
}
__device__ __forceinline__ float silu_f(float v) { return v / (1.0f + __expf(-v)); }

// ---- WMMA fragment helpers (CDNA5 16-bit layouts, wave32) -------------------
// A: 16x32 (MxK). Lane's 8 dwords form two contiguous 16B runs -> 2x ds_load_b128.
__device__ __forceinline__ v16bf load_A16x32(const unsigned short* base, int stride, int lane) {
    int m  = lane & 15;
    int hi = (lane >> 4) & 1;
    const unsigned short* rowp = base + m * stride;
    U8x32 s;
#pragma unroll
    for (int v = 0; v < 8; ++v) {
        int k = (v < 4) ? ((hi ? 8 : 0) + 2 * v) : ((hi ? 24 : 16) + 2 * (v - 4));
        s.u[v] = (unsigned)rowp[k] | ((unsigned)rowp[k + 1] << 16);
    }
    return __builtin_bit_cast(v16bf, s);
}

// B: 32x16 (KxN), pre-swizzled in global so each lane reads 8 contiguous dwords
// (two clause'd global_load_b128).
__device__ __forceinline__ v16bf load_Bsw(const unsigned int* __restrict__ W,
                                          int kt, int nt, int lane) {
    const unsigned int* p = W + ((((size_t)kt * 8) + nt) * 32 + lane) * 8;
    U8x32 s;
#pragma unroll
    for (int v = 0; v < 8; ++v) s.u[v] = p[v];
    return __builtin_bit_cast(v16bf, s);
}

__device__ __forceinline__ v8f wmma_bf(v16bf a, v16bf b, v8f c) {
    return __builtin_amdgcn_wmma_f32_16x16x32_bf16(false, a, false, b, (short)0, c, false, false);
}

// 2-deep software pipeline over the 8 N-tiles: the load of B[t+1] is issued
// before the WMMA on B[t], forcing two live B fragments so the backend can
// overlap B loads with matrix-pipe execution instead of draining loadcnt to 0.
__device__ __forceinline__ void mma_row8(v8f (&acc)[8], v16bf a,
                                         const unsigned int* __restrict__ W,
                                         int kt, int lane) {
    v16bf bb[2];
    bb[0] = load_Bsw(W, kt, 0, lane);
#pragma unroll
    for (int t = 0; t < 8; ++t) {
        if (t + 1 < 8) bb[(t + 1) & 1] = load_Bsw(W, kt, t + 1, lane);
        acc[t] = wmma_bf(a, bb[t & 1], acc[t]);
    }
}

__device__ __forceinline__ void acc_zero(v8f (&acc)[8]) {
#pragma unroll
    for (int t = 0; t < 8; ++t)
#pragma unroll
        for (int r = 0; r < 8; ++r) acc[t][r] = 0.0f;
}

// ---- small utility kernels --------------------------------------------------
__global__ void zero_kernel(float* p, size_t n) {
    size_t i = (size_t)blockIdx.x * blockDim.x + threadIdx.x;
    if (i < n) p[i] = 0.0f;
}

__global__ void f32_to_bf16_kernel(const float* __restrict__ s, unsigned short* __restrict__ d, size_t n) {
    size_t i = (size_t)blockIdx.x * blockDim.x + threadIdx.x;
    if (i < n) d[i] = f2bf(s[i]);
}

// Convert fp32 weight [K,128] into swizzled bf16 fragment layout, K padded to Kp.
__global__ void wswz_kernel(const float* __restrict__ W, unsigned int* __restrict__ dst, int K, int Kp) {
    int i = blockIdx.x * blockDim.x + threadIdx.x;
    int total = (Kp >> 5) * 8 * 32 * 8;
    if (i >= total) return;
    int v    = i & 7;
    int lane = (i >> 3) & 31;
    int nt   = (i >> 8) & 7;
    int kt   = i >> 11;
    int n = nt * 16 + (lane & 15);
    int k = kt * 32 + ((v < 4) ? (((lane & 16) ? 8 : 0) + 2 * v) : (((lane & 16) ? 24 : 16) + 2 * (v - 4)));
    unsigned int lo = (k     < K) ? f2bf(W[(size_t)k       * HD + n]) : 0u;
    unsigned int hi = (k + 1 < K) ? f2bf(W[(size_t)(k + 1) * HD + n]) : 0u;
    dst[i] = lo | (hi << 16);
}

// ---- pooling ---------------------------------------------------------------
__global__ void atom_scatter_kernel(const float* __restrict__ Hin, const float* __restrict__ Zin,
                                    const int* __restrict__ bid,
                                    float* __restrict__ hsum, float* __restrict__ xsum,
                                    float* __restrict__ cnt, int nAtoms) {
    size_t i = (size_t)blockIdx.x * blockDim.x + threadIdx.x;
    if (i >= (size_t)nAtoms * HD) return;
    int a = (int)(i >> 7), c = (int)(i & 127);
    int b = bid[a];
    atomicAdd(&hsum[(size_t)b * HD + c], Hin[i]);
    if (c < 3) atomicAdd(&xsum[(size_t)b * 3 + c], Zin[(size_t)a * 3 + c]);
    if (c == 0) atomicAdd(&cnt[b], 1.0f);
}

__global__ void pool_fin_kernel(float* __restrict__ hsum, float* __restrict__ xsum,
                                const float* __restrict__ cnt, float* __restrict__ HmOut, int B) {
    size_t i = (size_t)blockIdx.x * blockDim.x + threadIdx.x;
    if (i >= (size_t)B * HD) return;
    int b = (int)(i >> 7), c = (int)(i & 127);
    float cc = fmaxf(cnt[b], 1.0f);
    float v = hsum[i] / cc;
    hsum[i] = v;
    HmOut[i] = v;
    if (c < 3) xsum[(size_t)b * 3 + c] /= cc;
}

__global__ void edge_count_kernel(const int* __restrict__ erow, float* __restrict__ rcnt, int nE) {
    int i = blockIdx.x * blockDim.x + threadIdx.x;
    if (i < nE) atomicAdd(&rcnt[erow[i]], 1.0f);
}

__global__ void x_update_kernel(float* __restrict__ x, const float* __restrict__ xacc,
                                const float* __restrict__ rcnt, int B) {
    int i = blockIdx.x * blockDim.x + threadIdx.x;
    if (i < B * 3) x[i] += xacc[i] / fmaxf(rcnt[i / 3], 1.0f);
}

// ---- generic WMMA GEMM: out[M,128] = act(concat(src0,src1)@Wsw + bias) (+residual)
// Ping-pong LDS A staging: stage tile kt+1 while WMMAs consume tile kt.
__global__ void gemm_kernel(const float* __restrict__ src0, int k0,
                            const float* __restrict__ src1, int k1,
                            const unsigned int* __restrict__ Wsw,
                            const float* __restrict__ bias,
                            const float* __restrict__ residual,
                            float* __restrict__ out, int M, int Kp, int act) {
    __shared__ unsigned short As[2][64 * 32];
    const int tid = threadIdx.x, lane = tid & 31, wave = tid >> 5;
    const int rowBase = blockIdx.x * 64;

    auto stage = [&](int buf, int kt) {
        for (int idx = tid; idx < 64 * 32; idx += 128) {
            int r = idx >> 5, c = idx & 31;
            int row = rowBase + r, cg = (kt << 5) + c;
            float val = 0.0f;
            if (row < M) {
                if (cg < k0)           val = src0[(size_t)row * k0 + cg];
                else if (cg < k0 + k1) val = src1[(size_t)row * k1 + (cg - k0)];
            }
            As[buf][idx] = f2bf(val);
        }
    };

    v8f acc[8];
    acc_zero(acc);

    const int nK = Kp >> 5;
    stage(0, 0);
    __syncthreads();
    int p = 0;
    for (int kt = 0; kt < nK; ++kt) {
        if (kt + 1 < nK) stage(p ^ 1, kt + 1);
        v16bf a = load_A16x32(As[p] + wave * 16 * 32, 32, lane);
        mma_row8(acc, a, Wsw, kt, lane);
        __syncthreads();
        p ^= 1;
    }

    int hi = (lane >> 4) & 1, nlo = lane & 15;
#pragma unroll
    for (int t = 0; t < 8; ++t) {
        int col = t * 16 + nlo;
        float bv = bias ? bias[col] : 0.0f;
#pragma unroll
        for (int r = 0; r < 8; ++r) {
            int row = rowBase + wave * 16 + r + hi * 8;
            if (row < M) {
                float v = acc[t][r] + bv;
                if (act == 1) v = silu_f(v);
                if (residual) v += residual[(size_t)row * HD + col];
                out[(size_t)row * HD + col] = v;
            }
        }
    }
}

// ---- fused edge-MLP kernel: 64 edges per 4-wave workgroup -------------------
__global__ void edge_kernel(const float* __restrict__ x, const unsigned short* __restrict__ hbf,
                            const int* __restrict__ erow, const int* __restrict__ ecol,
                            const float* __restrict__ eattr,
                            const unsigned int* __restrict__ w1, const float* __restrict__ b1,
                            const unsigned int* __restrict__ w2, const float* __restrict__ b2,
                            const unsigned int* __restrict__ c1, const float* __restrict__ cb1,
                            const float* __restrict__ cw3,
                            float* __restrict__ agg, float* __restrict__ xacc, int nE) {
    __shared__ unsigned short As[2][64 * 32];
    __shared__ unsigned short T1[64 * HD];
    __shared__ unsigned short T2[64 * HD];
    __shared__ float Dif[64][3];
    __shared__ float Rad[64];
    __shared__ int   Rw[64], Cl[64];

    const int tid = threadIdx.x, lane = tid & 31, wave = tid >> 5;
    const int e0 = blockIdx.x * 64;
    const int hi = (lane >> 4) & 1, nlo = lane & 15;

    if (tid < 64) {
        int eg = e0 + tid;
        int r = 0, c = 0;
        float d0 = 0, d1 = 0, d2 = 0;
        if (eg < nE) {
            r = erow[eg]; c = ecol[eg];
            d0 = x[(size_t)r * 3 + 0] - x[(size_t)c * 3 + 0];
            d1 = x[(size_t)r * 3 + 1] - x[(size_t)c * 3 + 1];
            d2 = x[(size_t)r * 3 + 2] - x[(size_t)c * 3 + 2];
        }
        Rw[tid] = (eg < nE) ? r : -1;
        Cl[tid] = c;
        Dif[tid][0] = d0; Dif[tid][1] = d1; Dif[tid][2] = d2;
        Rad[tid] = d0 * d0 + d1 * d1 + d2 * d2;
    }
    __syncthreads();

    // gather/convert one 64x32 slice of e_in = [h[row] | h[col] | radial | eattr | 0pad]
    auto stage = [&](int buf, int kt) {
        for (int idx = tid; idx < 64 * 32; idx += 128) {
            int r = idx >> 5, c = idx & 31;
            int cg = (kt << 5) + c;
            unsigned short v = 0;
            int eg = e0 + r;
            if (eg < nE) {
                if (cg < 128)       v = hbf[(size_t)Rw[r] * HD + cg];
                else if (cg < 256)  v = hbf[(size_t)Cl[r] * HD + (cg - 128)];
                else if (cg == 256) v = f2bf(Rad[r]);
                else if (cg < 273)  v = f2bf(eattr[(size_t)eg * 16 + (cg - 257)]);
            }
            As[buf][idx] = v;
        }
    };

    v8f acc[8];

    // GEMM1: e_in[64,288] @ edge_w1 -> silu -> T1   (K padded 273->288)
    acc_zero(acc);
    stage(0, 0);
    __syncthreads();
    int p = 0;
    for (int kt = 0; kt < 9; ++kt) {
        if (kt + 1 < 9) stage(p ^ 1, kt + 1);
        v16bf a = load_A16x32(As[p] + wave * 16 * 32, 32, lane);
        mma_row8(acc, a, w1, kt, lane);
        __syncthreads();
        p ^= 1;
    }
#pragma unroll
    for (int t = 0; t < 8; ++t) {
        int col = t * 16 + nlo;
        float bv = b1[col];
#pragma unroll
        for (int r = 0; r < 8; ++r) {
            int m = wave * 16 + r + hi * 8;
            T1[m * HD + col] = f2bf(silu_f(acc[t][r] + bv));
        }
    }
    __syncthreads();

    // GEMM2: T1 @ edge_w2 -> silu -> m : store T2 (bf16) + atomicAdd agg (fp32)
    acc_zero(acc);
#pragma unroll
    for (int kt = 0; kt < 4; ++kt) {
        v16bf a = load_A16x32(T1 + wave * 16 * HD + kt * 32, HD, lane);
        mma_row8(acc, a, w2, kt, lane);
    }
#pragma unroll
    for (int t = 0; t < 8; ++t) {
        int col = t * 16 + nlo;
        float bv = b2[col];
#pragma unroll
        for (int r = 0; r < 8; ++r) {
            int m = wave * 16 + r + hi * 8;
            float v = silu_f(acc[t][r] + bv);
            T2[m * HD + col] = f2bf(v);
            int rg = Rw[m];
            if (rg >= 0) atomicAdd(&agg[(size_t)rg * HD + col], v);
        }
    }
    __syncthreads();

    // GEMM3: m @ coord_w1 -> silu -> T1 (reused)
    acc_zero(acc);
#pragma unroll
    for (int kt = 0; kt < 4; ++kt) {
        v16bf a = load_A16x32(T2 + wave * 16 * HD + kt * 32, HD, lane);
        mma_row8(acc, a, c1, kt, lane);
    }
#pragma unroll
    for (int t = 0; t < 8; ++t) {
        int col = t * 16 + nlo;
        float bv = cb1[col];
#pragma unroll
        for (int r = 0; r < 8; ++r) {
            int m = wave * 16 + r + hi * 8;
            T1[m * HD + col] = f2bf(silu_f(acc[t][r] + bv));
        }
    }
    __syncthreads();

    // per-edge scalar weight w = t2 . coord_w3 ; scatter diff*w into xacc
    if (tid < 64 && Rw[tid] >= 0) {
        float w = 0.0f;
#pragma unroll 4
        for (int j = 0; j < HD; ++j) w += bf2f(T1[tid * HD + j]) * cw3[j];
        int rg = Rw[tid];
        atomicAdd(&xacc[(size_t)rg * 3 + 0], Dif[tid][0] * w);
        atomicAdd(&xacc[(size_t)rg * 3 + 1], Dif[tid][1] * w);
        atomicAdd(&xacc[(size_t)rg * 3 + 2], Dif[tid][2] * w);
    }
}

// ---- normalization ----------------------------------------------------------
__global__ void norm_block_kernel(const float* __restrict__ hin, const int* __restrict__ gid,
                                  float* __restrict__ outBlock, float* __restrict__ gsum, int B) {
    __shared__ float red[128];
    int b = blockIdx.x, tid = threadIdx.x;
    float v = hin[(size_t)b * HD + tid];
    red[tid] = v * v;
    __syncthreads();
    for (int s = 64; s > 0; s >>= 1) {
        if (tid < s) red[tid] += red[tid + s];
        __syncthreads();
    }
    float n = fmaxf(sqrtf(red[0]), 1e-12f);
    float o = v / n;
    outBlock[(size_t)b * HD + tid] = o;
    atomicAdd(&gsum[(size_t)gid[b] * HD + tid], o);
}

__global__ void norm_graph_kernel(const float* __restrict__ gsum, float* __restrict__ outG) {
    __shared__ float red[128];
    int g = blockIdx.x, tid = threadIdx.x;
    float v = gsum[(size_t)g * HD + tid];
    red[tid] = v * v;
    __syncthreads();
    for (int s = 64; s > 0; s >>= 1) {
        if (tid < s) red[tid] += red[tid + s];
        __syncthreads();
    }
    float n = fmaxf(sqrtf(red[0]), 1e-12f);
    outG[(size_t)g * HD + tid] = v / n;
}

// ---- host orchestration -----------------------------------------------------
extern "C" void kernel_launch(void* const* d_in, const int* in_sizes, int n_in,
                              void* d_out, int out_size, void* d_ws, size_t ws_size,
                              hipStream_t stream) {
    const float* Hin      = (const float*)d_in[0];
    const float* Zin      = (const float*)d_in[1];
    const int*   bid      = (const int*)d_in[2];
    const int*   gid      = (const int*)d_in[3];
    const int*   edges    = (const int*)d_in[4];
    const float* eattr    = (const float*)d_in[5];
    const float* emb_in_w  = (const float*)d_in[6];
    const float* emb_in_b  = (const float*)d_in[7];
    const float* emb_out_w = (const float*)d_in[8];
    const float* emb_out_b = (const float*)d_in[9];
    const float* edge_w1  = (const float*)d_in[10];
    const float* edge_b1  = (const float*)d_in[11];
    const float* edge_w2  = (const float*)d_in[12];
    const float* edge_b2  = (const float*)d_in[13];
    const float* node_w1  = (const float*)d_in[14];
    const float* node_b1  = (const float*)d_in[15];
    const float* node_w2  = (const float*)d_in[16];
    const float* node_b2  = (const float*)d_in[17];
    const float* coord_w1 = (const float*)d_in[18];
    const float* coord_b1 = (const float*)d_in[19];
    const float* coord_w3 = (const float*)d_in[20];

    float* out = (float*)d_out;
    const int* erow = edges;
    const int* ecol = edges + NE;

    // workspace layout
    float* ws = (float*)d_ws;
    size_t off = 0;
    auto falloc = [&](size_t n) { float* p = ws + off; off += n; return p; };
    float* Hm   = falloc((size_t)NB * HD);
    float* hbuf = falloc((size_t)NB * HD);
    float* htmp = falloc((size_t)NB * HD);
    float* agg  = falloc((size_t)NB * HD);
    float* xpos = falloc((size_t)NB * 3);
    float* xacc = falloc((size_t)NB * 3);
    float* cnt  = falloc(NB);
    float* rcnt = falloc(NB);
    float* gsum = falloc((size_t)NG * HD);
    unsigned short* hbf = (unsigned short*)falloc((size_t)NB * HD / 2);
    unsigned int* wbuf = (unsigned int*)(ws + off);
    size_t woff = 0;
    auto walloc = [&](size_t n) { unsigned int* p = wbuf + woff; woff += n; return p; };
    unsigned int* w_ein = walloc(8192);
    unsigned int *w_e1[NL], *w_e2[NL], *w_c1[NL], *w_n1[NL], *w_n2[NL];
    for (int l = 0; l < NL; ++l) {
        w_e1[l] = walloc(18432);  // Kp=288
        w_e2[l] = walloc(8192);   // Kp=128
        w_c1[l] = walloc(8192);
        w_n1[l] = walloc(16384);  // Kp=256
        w_n2[l] = walloc(8192);
    }
    unsigned int* w_eout = walloc(8192);

    auto zblocks = [](size_t n) { return dim3((unsigned)((n + 255) / 256)); };

    // zero accumulators (fresh every launch)
    zero_kernel<<<zblocks((size_t)NB * HD), 256, 0, stream>>>(Hm, (size_t)NB * HD);
    zero_kernel<<<zblocks((size_t)NB * 3), 256, 0, stream>>>(xpos, (size_t)NB * 3);
    zero_kernel<<<zblocks(NB), 256, 0, stream>>>(cnt, NB);
    zero_kernel<<<zblocks(NB), 256, 0, stream>>>(rcnt, NB);
    zero_kernel<<<zblocks((size_t)NG * HD), 256, 0, stream>>>(gsum, (size_t)NG * HD);

    // atom -> block pooling
    atom_scatter_kernel<<<zblocks((size_t)NA * HD), 256, 0, stream>>>(Hin, Zin, bid, Hm, xpos, cnt, NA);
    pool_fin_kernel<<<zblocks((size_t)NB * HD), 256, 0, stream>>>(Hm, xpos, cnt, out, NB);
    edge_count_kernel<<<zblocks(NE), 256, 0, stream>>>(erow, rcnt, NE);

    // weight conversion to swizzled bf16 fragments
    wswz_kernel<<<zblocks(8192), 256, 0, stream>>>(emb_in_w, w_ein, 128, 128);
    for (int l = 0; l < NL; ++l) {
        wswz_kernel<<<zblocks(18432), 256, 0, stream>>>(edge_w1 + (size_t)l * 273 * HD, w_e1[l], 273, 288);
        wswz_kernel<<<zblocks(8192),  256, 0, stream>>>(edge_w2 + (size_t)l * HD * HD,  w_e2[l], 128, 128);
        wswz_kernel<<<zblocks(8192),  256, 0, stream>>>(coord_w1 + (size_t)l * HD * HD, w_c1[l], 128, 128);
        wswz_kernel<<<zblocks(16384), 256, 0, stream>>>(node_w1 + (size_t)l * 256 * HD, w_n1[l], 256, 256);
        wswz_kernel<<<zblocks(8192),  256, 0, stream>>>(node_w2 + (size_t)l * HD * HD,  w_n2[l], 128, 128);
    }
    wswz_kernel<<<zblocks(8192), 256, 0, stream>>>(emb_out_w, w_eout, 128, 128);

    // h = Hm @ emb_in_w + b
    gemm_kernel<<<dim3((NB + 63) / 64), 128, 0, stream>>>(Hm, 128, (const float*)nullptr, 0,
                                                          w_ein, emb_in_b, (const float*)nullptr,
                                                          hbuf, NB, 128, 0);

    for (int l = 0; l < NL; ++l) {
        f32_to_bf16_kernel<<<zblocks((size_t)NB * HD), 256, 0, stream>>>(hbuf, hbf, (size_t)NB * HD);
        zero_kernel<<<zblocks((size_t)NB * HD), 256, 0, stream>>>(agg, (size_t)NB * HD);
        zero_kernel<<<zblocks((size_t)NB * 3), 256, 0, stream>>>(xacc, (size_t)NB * 3);

        edge_kernel<<<dim3(NE / 64), 128, 0, stream>>>(xpos, hbf, erow, ecol, eattr,
                                                       w_e1[l], edge_b1 + l * HD,
                                                       w_e2[l], edge_b2 + l * HD,
                                                       w_c1[l], coord_b1 + l * HD,
                                                       coord_w3 + l * HD,
                                                       agg, xacc, NE);

        x_update_kernel<<<zblocks((size_t)NB * 3), 256, 0, stream>>>(xpos, xacc, rcnt, NB);

        // node MLP: htmp = silu(concat(h, agg) @ node_w1 + b1) ; h += htmp @ node_w2 + b2
        gemm_kernel<<<dim3((NB + 63) / 64), 128, 0, stream>>>(hbuf, 128, agg, 128,
                                                              w_n1[l], node_b1 + l * HD,
                                                              (const float*)nullptr, htmp, NB, 256, 1);
        gemm_kernel<<<dim3((NB + 63) / 64), 128, 0, stream>>>(htmp, 128, (const float*)nullptr, 0,
                                                              w_n2[l], node_b2 + l * HD,
                                                              hbuf, hbuf, NB, 128, 0);
    }

    // emb_out + normalizations
    gemm_kernel<<<dim3((NB + 63) / 64), 128, 0, stream>>>(hbuf, 128, (const float*)nullptr, 0,
                                                          w_eout, emb_out_b, (const float*)nullptr,
                                                          htmp, NB, 128, 0);
    norm_block_kernel<<<dim3(NB), 128, 0, stream>>>(htmp, gid, out + (size_t)NB * HD, gsum, NB);
    norm_graph_kernel<<<dim3(NG), 128, 0, stream>>>(gsum, out + 2 * (size_t)NB * HD);
}
